// MoERouter_52888227283709
// MI455X (gfx1250) — matrix operand order repaired and checked by reading, not compiled, not run on hardware
//
#include <hip/hip_runtime.h>
#include <hip/hip_bf16.h>

// ---------------- problem constants (from reference) ----------------
#define D_MODEL          4096
#define N_EXPERTS        64
#define TOKENS_TOTAL     (4 * 8192)
#define TOP_K            2

// ---------------- tiling ----------------
#define TOKENS_PER_WAVE  16                      // M of one WMMA tile
#define WAVES_PER_BLOCK  8                       // 256 threads, wave32
#define TOKENS_PER_BLOCK (TOKENS_PER_WAVE * WAVES_PER_BLOCK)
#define N_TILES          (N_EXPERTS / 16)        // 4 expert tiles of 16

typedef __attribute__((ext_vector_type(16))) __bf16 v16bf;
typedef __attribute__((ext_vector_type(8)))  __bf16 v8bf;
typedef __attribute__((ext_vector_type(8)))  float  v8f;

// One-shot conversion of W (fp32, [64 x 4096]) -> bf16 scratch. 1 MB read,
// 512 KB written; W then stays L2-resident for the router kernel.
__global__ void __launch_bounds__(256)
cvt_w_bf16_kernel(const float* __restrict__ W, __bf16* __restrict__ Wb, int n) {
  int i = blockIdx.x * blockDim.x + threadIdx.x;
  if (i < n) Wb[i] = (__bf16)W[i];
}

// Router: logits = x @ W^T via v_wmma_f32_16x16x32_bf16, then per-token top-2
// + 2-way softmax. Each wave: 16 tokens x 64 experts, K-loop over 4096.
template <bool WB16>
__global__ void __launch_bounds__(256)
moe_router_kernel(const float*  __restrict__ x,
                  const float*  __restrict__ Wf,   // fp32 W (fallback path)
                  const __bf16* __restrict__ Wb,   // bf16 W (fast path)
                  float* __restrict__ outW,
                  int*   __restrict__ outE) {
  const int lane = threadIdx.x & 31;
  const int wave = threadIdx.x >> 5;
  const int row  = lane & 15;   // M (token-in-tile) for A; N (expert-in-tile) for B
  const int hi   = lane >> 4;   // selects K sub-chunks per the wave32 WMMA layouts

  const size_t tok0 = (size_t)blockIdx.x * TOKENS_PER_BLOCK
                    + (size_t)wave * TOKENS_PER_WAVE;
  const float* A = x + (tok0 + (size_t)row) * D_MODEL;

  v8f c[N_TILES] = {};   // 4 x (16x16 f32 accumulator, 8 VGPRs each)

  for (int k = 0; k < D_MODEL; k += 32) {
    // Branch-free WGP-scope prefetch of the x stream ~16 iterations ahead.
    // Masked into the current row so the address is always valid (WGP-scope
    // prefetches are non-speculative per ISA 10.5).
    __builtin_prefetch(A + ((k + 512) & (D_MODEL - 1)), 0, 3);

    // ---- A fragment: 16x32 tile in this code's own k order (B uses the same).
    // lanes 0-15 hold K 0..15; lanes 16-31 hold K 16..31 (contiguous per lane).
    v8f fa0 = *(const v8f*)(A + k + hi * 16);
    v8f fa1 = *(const v8f*)(A + k + hi * 16 + 8);
    v8bf alo = __builtin_convertvector(fa0, v8bf);
    v8bf ahi = __builtin_convertvector(fa1, v8bf);
    v16bf a = __builtin_shufflevector(alo, ahi,
        0, 1, 2, 3, 4, 5, 6, 7, 8, 9, 10, 11, 12, 13, 14, 15);

    // ---- B fragments: 32x16 tile in the same k order as A.
    // lanes 0-15 hold K 0..15; lanes 16-31 hold K 16..31 (contiguous per lane).
#pragma unroll
    for (int t = 0; t < N_TILES; ++t) {
      const size_t expert = (size_t)(t * 16 + row);
      v16bf b;
      if constexpr (WB16) {
        b = *(const v16bf*)(Wb + expert * D_MODEL + k + hi * 16);
      } else {
        v8f fb0 = *(const v8f*)(Wf + expert * D_MODEL + k + hi * 16);
        v8f fb1 = *(const v8f*)(Wf + expert * D_MODEL + k + hi * 16 + 8);
        v8bf blo = __builtin_convertvector(fb0, v8bf);
        v8bf bhi = __builtin_convertvector(fb1, v8bf);
        b = __builtin_shufflevector(blo, bhi,
            0, 1, 2, 3, 4, 5, 6, 7, 8, 9, 10, 11, 12, 13, 14, 15);
      }
      c[t] = __builtin_amdgcn_wmma_f32_16x16x32_bf16(
          /*neg_a=*/false, a, /*neg_b=*/false, b,
          /*c_mod=*/(short)0, c[t], /*reuse_a=*/false, /*reuse_b=*/false);
    }
  }

  // ---- Spill logits to LDS: [wave][token 0..15][expert 0..63].
  // C/D layout: VGPR r, lanes 0-15 -> M=r; lanes 16-31 -> M=r+8; N=lane&15.
  __shared__ float lds[WAVES_PER_BLOCK][TOKENS_PER_WAVE][N_EXPERTS];
#pragma unroll
  for (int t = 0; t < N_TILES; ++t)
#pragma unroll
    for (int r = 0; r < 8; ++r)
      lds[wave][r + hi * 8][t * 16 + row] = c[t][r];

  __syncthreads();

  // ---- Top-2 + softmax: one lane per token.
  if (lane < TOKENS_PER_WAVE) {
    const float* lrow = lds[wave][lane];
    float v1 = -3.4e38f, v2 = -3.4e38f;
    int   i1 = 0, i2 = 0;
#pragma unroll 8
    for (int e = 0; e < N_EXPERTS; ++e) {
      float v = lrow[e];
      if (v > v1)      { v2 = v1; i2 = i1; v1 = v; i1 = e; }  // strict '>' ==
      else if (v > v2) { v2 = v;  i2 = e; }                   // jax tie-break
    }
    // softmax over {v1, v2} with v1 >= v2 (numerically stable).
    float e2 = __expf(v2 - v1);
    float s  = 1.0f + e2;
    size_t tok = tok0 + (size_t)lane;
    outW[tok * TOP_K + 0] = 1.0f / s;
    outW[tok * TOP_K + 1] = e2 / s;
    outE[tok * TOP_K + 0] = i1;
    outE[tok * TOP_K + 1] = i2;
  }
}

extern "C" void kernel_launch(void* const* d_in, const int* in_sizes, int n_in,
                              void* d_out, int out_size, void* d_ws, size_t ws_size,
                              hipStream_t stream) {
  const float* x = (const float*)d_in[0];                 // [4, 8192, 4096] f32
  const float* W = (const float*)d_in[1];                 // [64, 4096] f32

  float* outW = (float*)d_out;                            // [32768, 2] f32
  int*   outE = (int*)d_out + (size_t)TOKENS_TOTAL * TOP_K; // [32768, 2] i32

  const size_t wb_bytes = (size_t)N_EXPERTS * D_MODEL * sizeof(__bf16);
  dim3 grid(TOKENS_TOTAL / TOKENS_PER_BLOCK);             // 256 blocks
  dim3 block(32 * WAVES_PER_BLOCK);                       // 256 threads

  if (ws_size >= wb_bytes) {
    __bf16* Wb = (__bf16*)d_ws;
    int n = N_EXPERTS * D_MODEL;
    cvt_w_bf16_kernel<<<(n + 255) / 256, 256, 0, stream>>>(W, Wb, n);
    moe_router_kernel<true><<<grid, block, 0, stream>>>(x, W, Wb, outW, outE);
  } else {
    moe_router_kernel<false><<<grid, block, 0, stream>>>(x, W, nullptr, outW, outE);
  }
}
